// Resnet50ContextualLoss_45775761441353
// MI455X (gfx1250) — compile-verified, matrix operand order checked by reading, not run on hardware
//
#include <hip/hip_runtime.h>
#include <hip/hip_bf16.h>

// ---------------------------------------------------------------------------
// Contextual loss, fully fused for gfx1250 (MI455X).
//   N=4, C=256, H=W=64 -> P=4096
//   cosine_sim: 4 batched GEMMs (4096x256)^T * (256x4096) via
//   v_wmma_f32_16x16x32_f16, one 16-row strip per workgroup kept in LDS
//   (16 x 4096 f16 = 128KB of the 320KB WGP LDS); the 256MB similarity
//   tensor never touches HBM.
// ---------------------------------------------------------------------------

#define NN 4
#define CC 256
#define PP 4096           // H*W
#define PITCH 4100        // LDS row pitch in halves (bank-conflict skew)
#define EPS_D 1e-05f
#define EPS_LOG 1e-05f
#define INV_H 2.0f        // 1 / H_BAND, H_BAND = 0.5

typedef _Float16 v8h  __attribute__((ext_vector_type(8)));
typedef _Float16 v16h __attribute__((ext_vector_type(16)));
typedef float    v8f  __attribute__((ext_vector_type(8)));

// ---------------------------------------------------------------- zero colmax
__global__ __launch_bounds__(256) void zero_kernel(float* __restrict__ p, int n) {
  int i = blockIdx.x * 256 + threadIdx.x;
  if (i < n) p[i] = 0.0f;
}

// ------------------------------------------------- per-channel mean of y_hat
__global__ __launch_bounds__(256) void mean_kernel(const float* __restrict__ yh,
                                                   float* __restrict__ mu) {
  __shared__ float red[256];
  const int c = blockIdx.x;
  const int tid = threadIdx.x;
  float s = 0.0f;
  for (int i = tid; i < NN * PP; i += 256) {
    int n = i >> 12, p = i & (PP - 1);
    s += yh[(((size_t)n * CC + c) << 12) + p];
  }
  red[tid] = s;
  __syncthreads();
  for (int off = 128; off > 0; off >>= 1) {
    if (tid < off) red[tid] += red[tid + off];
    __syncthreads();
  }
  if (tid == 0) mu[c] = red[0] * (1.0f / (NN * PP));
}

// ---- center by mu, L2-normalize over C, emit f16 in (n, p, c) WMMA layout
__global__ __launch_bounds__(256) void normalize_kernel(const float* __restrict__ x,
                                                        const float* __restrict__ yh,
                                                        const float* __restrict__ mu,
                                                        _Float16* __restrict__ xn,
                                                        _Float16* __restrict__ yn) {
  __shared__ float mu_s[CC];
  const int tid = threadIdx.x;
  mu_s[tid] = mu[tid];
  __syncthreads();

  const int g = blockIdx.x * 256 + tid;   // one (n, p) per thread
  const int n = g >> 12;
  const int p = g & (PP - 1);
  const float* xr = x  + ((size_t)n * CC) * PP + p;   // stride PP over c
  const float* yr = yh + ((size_t)n * CC) * PP + p;

  float sx = 0.0f, sy = 0.0f;
  for (int c = 0; c < CC; ++c) {
    float dx = xr[(size_t)c * PP] - mu_s[c];
    float dy = yr[(size_t)c * PP] - mu_s[c];
    sx += dx * dx;
    sy += dy * dy;
  }
  const float rnx = rsqrtf(sx + 1e-20f);
  const float rny = rsqrtf(sy + 1e-20f);

  _Float16* xo = xn + ((size_t)(n * PP + p)) * CC;
  _Float16* yo = yn + ((size_t)(n * PP + p)) * CC;
  for (int c0 = 0; c0 < CC; c0 += 8) {
    v8h vx, vy;
#pragma unroll
    for (int j = 0; j < 8; ++j) {
      vx[j] = (_Float16)((xr[(size_t)(c0 + j) * PP] - mu_s[c0 + j]) * rnx);
      vy[j] = (_Float16)((yr[(size_t)(c0 + j) * PP] - mu_s[c0 + j]) * rny);
    }
    *(v8h*)(xo + c0) = vx;
    *(v8h*)(yo + c0) = vy;
  }
}

// ------------------------------------------------------- fused GEMM + CX core
// One workgroup = one (n, 16-row p-strip). 8 waves; wave w owns q-tiles
// t = w, w+8, ... Strip of d = 1 - cos kept in LDS, then row-min, exp,
// row-sum, column-max -> global atomicMax.
__global__ __launch_bounds__(256) void cx_kernel(const _Float16* __restrict__ xn,
                                                 const _Float16* __restrict__ yn,
                                                 float* __restrict__ colmax) {
  extern __shared__ char smem[];
  _Float16* wbuf   = (_Float16*)smem;                               // 16 x PITCH
  float*    red    = (float*)(smem + 16 * PITCH * sizeof(_Float16)); // 256
  float*    rowmin = red + 256;                                      // 16
  float*    rowscl = rowmin + 16;                                    // 16

  const int tid   = threadIdx.x;
  const int lane  = tid & 31;
  const int wave  = tid >> 5;
  const int n     = blockIdx.x >> 8;
  const int strip = blockIdx.x & 255;
  const int pBase = strip << 4;
  const int lr    = lane & 15;   // row (A) / col (B,D) within tile
  const int hi    = lane >> 4;   // K-half selector

  // A strip: 16 p-rows x 256 K, resident for the whole block row.
  // 16-bit A layout: lane L holds M=L%16; VGPRs 0-3 = K {0..7}+8*hi,
  // VGPRs 4-7 = K {16..23}+8*hi (per 32-wide K step).
  const _Float16* Abase = xn + ((size_t)(n * PP + pBase + lr)) * CC;
  v16h a[8];
#pragma unroll
  for (int kb = 0; kb < 8; ++kb) {
    const int c0 = kb * 32 + hi * 8;
    v8h alo = *(const v8h*)(Abase + c0);
    v8h ahi = *(const v8h*)(Abase + c0 + 16);
    a[kb] = __builtin_shufflevector(alo, ahi, 0, 1, 2, 3, 4, 5, 6, 7,
                                    8, 9, 10, 11, 12, 13, 14, 15);
  }

  for (int t = wave; t < 256; t += 8) {
    const int qBase = t << 4;
    const _Float16* Bbase = yn + ((size_t)(n * PP + qBase + lr)) * CC;
    v16h b[8];
#pragma unroll
    for (int kb = 0; kb < 8; ++kb) {
      const int c0 = kb * 32 + hi * 8;
      v8h blo = *(const v8h*)(Bbase + c0);
      v8h bhi = *(const v8h*)(Bbase + c0 + 16);
      b[kb] = __builtin_shufflevector(blo, bhi, 0, 1, 2, 3, 4, 5, 6, 7,
                                      8, 9, 10, 11, 12, 13, 14, 15);
    }
    v8f acc = {};
#pragma unroll
    for (int kb = 0; kb < 8; ++kb)
      acc = __builtin_amdgcn_wmma_f32_16x16x32_f16(
          false, a[kb], false, b[kb], (short)0, acc, false, false);

    // d = 1 - cos; D layout: VGPR i -> row i + 8*hi, col = qBase + lr.
#pragma unroll
    for (int i = 0; i < 8; ++i) {
      const int r = i + 8 * hi;
      wbuf[r * PITCH + qBase + lr] = (_Float16)(1.0f - acc[i]);
    }
  }
  __syncthreads();

  // ---- row min (16 threads per row)
  const int rr  = tid >> 4;
  const int sub = tid & 15;
  float m = 3.4e38f;
  for (int j = sub; j < PP; j += 16)
    m = fminf(m, (float)wbuf[rr * PITCH + j]);
  red[rr * 16 + sub] = m;
  __syncthreads();
  if (sub == 0) {
    float mm = red[rr * 16];
#pragma unroll
    for (int k = 1; k < 16; ++k) mm = fminf(mm, red[rr * 16 + k]);
    rowmin[rr] = mm;
  }
  __syncthreads();

  // ---- w = exp((1 - d/(dmin+eps))/h), overwrite strip, accumulate row sum
  const float inv_dm = 1.0f / (rowmin[rr] + EPS_D);
  float s = 0.0f;
  for (int j = sub; j < PP; j += 16) {
    float d = (float)wbuf[rr * PITCH + j];
    float w = __expf((1.0f - d * inv_dm) * INV_H);
    wbuf[rr * PITCH + j] = (_Float16)w;
    s += w;
  }
  red[rr * 16 + sub] = s;
  __syncthreads();
  if (sub == 0) {
    float ss = 0.0f;
#pragma unroll
    for (int k = 0; k < 16; ++k) ss += red[rr * 16 + k];
    rowscl[rr] = 1.0f / ss;
  }
  __syncthreads();

  // ---- per-column max of cx_ij over this strip -> global atomic max.
  // cx_ij > 0, so positive-float bit pattern compares correctly as int.
  for (int j = 0; j < 16; ++j) {
    const int q = tid + (j << 8);
    float cm = 0.0f;
#pragma unroll
    for (int r = 0; r < 16; ++r)
      cm = fmaxf(cm, (float)wbuf[r * PITCH + q] * rowscl[r]);
    atomicMax((int*)(colmax + n * PP + q), __float_as_int(cm));
  }
}

// ---------------------------------------------- cx mean, -log, batch mean
__global__ __launch_bounds__(256) void finalize_kernel(const float* __restrict__ colmax,
                                                       float* __restrict__ out) {
  __shared__ float red[256];
  const int tid = threadIdx.x;
  float acc = 0.0f;
  for (int n = 0; n < NN; ++n) {
    float s = 0.0f;
    for (int q = tid; q < PP; q += 256) s += colmax[n * PP + q];
    red[tid] = s;
    __syncthreads();
    for (int off = 128; off > 0; off >>= 1) {
      if (tid < off) red[tid] += red[tid + off];
      __syncthreads();
    }
    if (tid == 0) acc += -logf(red[0] * (1.0f / PP) + EPS_LOG);
    __syncthreads();
  }
  if (tid == 0) out[0] = acc * (1.0f / NN);
}

// ---------------------------------------------------------------------------
extern "C" void kernel_launch(void* const* d_in, const int* in_sizes, int n_in,
                              void* d_out, int out_size, void* d_ws, size_t ws_size,
                              hipStream_t stream) {
  const float* y  = (const float*)d_in[0];   // "x" of contextual_loss
  const float* yh = (const float*)d_in[1];   // "y" of contextual_loss (mu source)
  float* out = (float*)d_out;

  // workspace carve: mu(1KB) | colmax(64KB) | xn f16(8MB) | yn f16(8MB)
  char* ws = (char*)d_ws;
  float*     mu     = (float*)ws;
  float*     colmax = (float*)(ws + 1024);
  _Float16*  xn     = (_Float16*)(ws + 1024 + (size_t)NN * PP * sizeof(float));
  _Float16*  yn     = xn + (size_t)NN * PP * CC;

  zero_kernel<<<(NN * PP + 255) / 256, 256, 0, stream>>>(colmax, NN * PP);
  mean_kernel<<<CC, 256, 0, stream>>>(yh, mu);
  normalize_kernel<<<(NN * PP) / 256, 256, 0, stream>>>(y, yh, mu, xn, yn);

  const size_t shbytes = (size_t)16 * PITCH * sizeof(_Float16) + (256 + 32) * sizeof(float);
  cx_kernel<<<NN * (PP / 16), 256, shbytes, stream>>>(xn, yn, colmax);

  finalize_kernel<<<1, 256, 0, stream>>>(colmax, out);
}
